// MultiHeadSelfAttention_10514079941420
// MI455X (gfx1250) — compile-verified, matrix operand order checked by reading, not run on hardware
//
#include <hip/hip_runtime.h>
#include <hip/hip_bf16.h>

typedef _Float16 half_t;
typedef __attribute__((ext_vector_type(16))) _Float16 v16h;
typedef __attribute__((ext_vector_type(8)))  _Float16 v8h;
typedef __attribute__((ext_vector_type(4)))  _Float16 v4h;
typedef __attribute__((ext_vector_type(8)))  float    v8f;
typedef __attribute__((ext_vector_type(4)))  float    v4f;

#define NB   4
#define SS   2048
#define DD   768
#define HH   12
#define HD   64
#define TOPK 512

// ---------------------------------------------------------------------------
// WMMA helper: D = A(16x32 f16) x B(32x16 f16) + C(16x16 f32)
// ---------------------------------------------------------------------------
__device__ inline v8f wmma16(v16h a, v16h b, v8f c) {
    return __builtin_amdgcn_wmma_f32_16x16x32_f16(
        /*neg_a=*/false, a, /*neg_b=*/false, b,
        /*c_mod=*/(short)0, c, /*reuse_a=*/false, /*reuse_b=*/false);
}

// Fragment load for NT layout (K contiguous). p already includes the per-lane
// row offset and hf*8 K offset. Elements e: K = (e/8)*16 + e%8 rel. to base.
__device__ inline v16h frag_f16(const half_t* p) {
    v8h lo = *(const v8h*)(p);
    v8h hi = *(const v8h*)(p + 16);
    return __builtin_shufflevector(lo, hi, 0, 1, 2, 3, 4, 5, 6, 7,
                                            8, 9, 10, 11, 12, 13, 14, 15);
}

// ---------------------------------------------------------------------------
// Kernel 1: convert the four DxD weight matrices to f16
// ---------------------------------------------------------------------------
__global__ void cvt_weights_kernel(const float* Wq, const float* Wk,
                                   const float* Wv, const float* Wo,
                                   half_t* q16, half_t* k16,
                                   half_t* v16, half_t* o16) {
    int i = blockIdx.x * blockDim.x + threadIdx.x;
    if (i < DD * DD) {
        q16[i] = (half_t)Wq[i];
        k16[i] = (half_t)Wk[i];
        v16[i] = (half_t)Wv[i];
        o16[i] = (half_t)Wo[i];
    }
}

// ---------------------------------------------------------------------------
// Kernel 2: deterministic compaction of uniform_set -> idx[512], and
// per-batch key bias from the padding mask (0 or -1e30). One block, no
// atomics -> bitwise deterministic across graph replays.
// ---------------------------------------------------------------------------
__global__ void build_idx_kernel(const unsigned char* uset, const float* mask,
                                 int* idx, float* keybias) {
    __shared__ int counts[257];
    int t = threadIdx.x;
    int cnt = 0;
#pragma unroll
    for (int i = 0; i < 8; i++) cnt += (uset[t * 8 + i] != 0) ? 1 : 0;
    counts[t + 1] = cnt;
    if (t == 0) counts[0] = 0;
    __syncthreads();
    if (t == 0) {
        for (int i = 1; i <= 256; i++) counts[i] += counts[i - 1];
    }
    __syncthreads();
    int pos = counts[t];
    for (int i = 0; i < 8; i++) {
        int g = t * 8 + i;
        if (uset[g]) idx[pos++] = g;
    }
    __syncthreads();
    for (int i = t; i < NB * TOPK; i += 256) {
        int b  = i >> 9;
        int jj = i & (TOPK - 1);
        float m = mask[b * SS + idx[jj]];
        keybias[i] = (m < 0.0f) ? -1e30f : 0.0f;
    }
}

// ---------------------------------------------------------------------------
// Kernel 3: row convert (+optional gather) f32 -> f16.
// grid.x = output rows, block = 192 threads (4 elems each, 768/row).
// ---------------------------------------------------------------------------
__global__ void cvt_rows_kernel(const float* __restrict__ src,
                                const int* __restrict__ idx,
                                half_t* __restrict__ dst, int use_gather) {
    int row = blockIdx.x;
    int srcrow;
    if (use_gather) {
        int b  = row >> 9;
        int jj = row & (TOPK - 1);
        srcrow = b * SS + idx[jj];
    } else {
        srcrow = row;
    }
    const float* p = src + (size_t)srcrow * DD + threadIdx.x * 4;
    v4f x = *(const v4f*)p;
    v4h y;
#pragma unroll
    for (int i = 0; i < 4; i++) y[i] = (half_t)x[i];
    *(v4h*)(dst + (size_t)row * DD + threadIdx.x * 4) = y;
}

// ---------------------------------------------------------------------------
// Kernels 4-6, 8: unified pure-f16 NT GEMM: out = (A @ W^T + bias) * scale
// A [M,768] f16 row-major, W [768,768] f16 row-major (rows = output cols).
// Block = 128 threads (4 waves); block tile 64x128; wave tile 32x64.
// One B fragment live at a time -> low register pressure, no spills; the
// compiler's scheduler overlaps the b128 loads with WMMA issue.
//   mode 0: Q layout [B,H,S,HD] f16 (scaled)
//   mode 1: K layout [B,H,TOPK,HD] f16
//   mode 2: V transposed [B,H,HD,TOPK] f16
//   mode 3: f32 output [M,768] (final projection)
// ---------------------------------------------------------------------------
__global__ void gemm_nt_kernel(const half_t* __restrict__ A,
                               const half_t* __restrict__ W,
                               const float* __restrict__ bias,
                               half_t* __restrict__ outh,
                               float* __restrict__ outf,
                               int mode, float scale) {
    const int lane = threadIdx.x & 31;
    const int w    = threadIdx.x >> 5;          // 0..3
    const int ln   = lane & 15;
    const int hf   = lane >> 4;
    const int wm   = (w & 1) * 32;
    const int wn   = (w >> 1) * 64;
    const int bm   = blockIdx.x * 64;
    const int bn   = blockIdx.y * 128;

    const half_t* arow[2];
#pragma unroll
    for (int mi = 0; mi < 2; mi++)
        arow[mi] = A + (size_t)(bm + wm + mi * 16 + ln) * DD + hf * 8;
    const half_t* brow[4];
#pragma unroll
    for (int ni = 0; ni < 4; ni++)
        brow[ni] = W + (size_t)(bn + wn + ni * 16 + ln) * DD + hf * 8;

    v8f c[2][4];
#pragma unroll
    for (int mi = 0; mi < 2; mi++)
#pragma unroll
        for (int ni = 0; ni < 4; ni++) c[mi][ni] = {};

    for (int k0 = 0; k0 < DD; k0 += 32) {
        v16h a0 = frag_f16(arow[0] + k0);
        v16h a1 = frag_f16(arow[1] + k0);
#pragma unroll
        for (int ni = 0; ni < 4; ni++) {
            v16h bf = frag_f16(brow[ni] + k0);
            c[0][ni] = wmma16(a0, bf, c[0][ni]);
            c[1][ni] = wmma16(a1, bf, c[1][ni]);
        }
    }

#pragma unroll
    for (int mi = 0; mi < 2; mi++) {
#pragma unroll
        for (int ni = 0; ni < 4; ni++) {
            int col = bn + wn + ni * 16 + ln;       // 0..767
            float bv = bias[col];
            int h  = col >> 6;
            int hd = col & 63;
#pragma unroll
            for (int v = 0; v < 8; v++) {
                int row = bm + wm + mi * 16 + v + 8 * hf;
                float val = (c[mi][ni][v] + bv) * scale;
                if (mode == 0) {
                    int b = row >> 11, s = row & (SS - 1);
                    outh[(((size_t)b * HH + h) * SS + s) * HD + hd] = (half_t)val;
                } else if (mode == 1) {
                    int b = row >> 9, jj = row & (TOPK - 1);
                    outh[(((size_t)b * HH + h) * TOPK + jj) * HD + hd] = (half_t)val;
                } else if (mode == 2) {
                    int b = row >> 9, jj = row & (TOPK - 1);
                    outh[(((size_t)b * HH + h) * HD + hd) * TOPK + jj] = (half_t)val;
                } else {
                    outf[(size_t)row * DD + col] = val;
                }
            }
        }
    }
}

// ---------------------------------------------------------------------------
// Kernel 7: attention over the 512 gathered keys.
// grid = (S/32, B*H), block = 256 (8 waves). Each block: 32 queries.
// Phase 1: scores (32x512) in registers; phase 2: softmax stats via shfl+LDS,
// exp'd probs (f16) to LDS; phase 3: ctx = P @ Vc via WMMA (Vc pre-transposed)
// with a global_prefetch stream hint on the V operand.
// ---------------------------------------------------------------------------
__global__ void attn_kernel(const half_t* __restrict__ Q16,
                            const half_t* __restrict__ Kc16,
                            const half_t* __restrict__ VcT,
                            const float* __restrict__ keybias,
                            half_t* __restrict__ ctx16) {
    __shared__ float  redbuf[8][32];
    __shared__ float  rowmax[32];
    __shared__ float  rowsum[32];
    __shared__ half_t probs[32][TOPK + 16];   // padded row stride

    const int lane = threadIdx.x & 31;
    const int w    = threadIdx.x >> 5;     // 0..7
    const int ln   = lane & 15;
    const int hf   = lane >> 4;
    const int bh   = blockIdx.y;           // 0..47
    const int b    = bh / HH;
    const int h    = bh - b * HH;
    const int q0   = blockIdx.x * 32;

    // ---- phase 1: score tiles st[2][4]: rows q0..q0+31, cols w*64..w*64+63
    const half_t* Qbase = Q16 + ((size_t)bh * SS + q0) * HD;
    const half_t* Kbase = Kc16 + (size_t)bh * TOPK * HD;

    v16h aq[2][2];
#pragma unroll
    for (int mi = 0; mi < 2; mi++) {
        const half_t* qp = Qbase + (size_t)(mi * 16 + ln) * HD + hf * 8;
#pragma unroll
        for (int kc = 0; kc < 2; kc++) aq[mi][kc] = frag_f16(qp + kc * 32);
    }

    v8f st[2][4];
#pragma unroll
    for (int mi = 0; mi < 2; mi++)
#pragma unroll
        for (int nt = 0; nt < 4; nt++) st[mi][nt] = {};

#pragma unroll
    for (int nt = 0; nt < 4; nt++) {
        const half_t* kp = Kbase + (size_t)(w * 64 + nt * 16 + ln) * HD + hf * 8;
        v16h b0 = frag_f16(kp);
        v16h b1 = frag_f16(kp + 32);
        st[0][nt] = wmma16(aq[0][0], b0, st[0][nt]);
        st[0][nt] = wmma16(aq[0][1], b1, st[0][nt]);
        st[1][nt] = wmma16(aq[1][0], b0, st[1][nt]);
        st[1][nt] = wmma16(aq[1][1], b1, st[1][nt]);
    }

    // add per-key bias (padding mask)
#pragma unroll
    for (int nt = 0; nt < 4; nt++) {
        int col = w * 64 + nt * 16 + ln;
        float kb = keybias[b * TOPK + col];
#pragma unroll
        for (int mi = 0; mi < 2; mi++)
#pragma unroll
            for (int v = 0; v < 8; v++) st[mi][nt][v] += kb;
    }

    // ---- phase 2a: per-wave row max
#pragma unroll
    for (int mi = 0; mi < 2; mi++) {
#pragma unroll
        for (int v = 0; v < 8; v++) {
            int r = mi * 16 + v + 8 * hf;
            float m = fmaxf(fmaxf(st[mi][0][v], st[mi][1][v]),
                            fmaxf(st[mi][2][v], st[mi][3][v]));
#pragma unroll
            for (int off = 1; off < 16; off <<= 1)
                m = fmaxf(m, __shfl_xor(m, off, 32));
            if (ln == 0) redbuf[w][r] = m;
        }
    }
    __syncthreads();
    if (threadIdx.x < 32) {
        float m = redbuf[0][threadIdx.x];
#pragma unroll
        for (int ww = 1; ww < 8; ww++) m = fmaxf(m, redbuf[ww][threadIdx.x]);
        rowmax[threadIdx.x] = m;
    }
    __syncthreads();

    // ---- phase 2b: exp, probs -> LDS, partial row sums
#pragma unroll
    for (int mi = 0; mi < 2; mi++) {
#pragma unroll
        for (int v = 0; v < 8; v++) {
            int r = mi * 16 + v + 8 * hf;
            float rm = rowmax[r];
            float s = 0.0f;
#pragma unroll
            for (int nt = 0; nt < 4; nt++) {
                int col = w * 64 + nt * 16 + ln;
                float e = __expf(st[mi][nt][v] - rm);
                probs[r][col] = (half_t)e;
                s += e;
            }
#pragma unroll
            for (int off = 1; off < 16; off <<= 1)
                s += __shfl_xor(s, off, 32);
            if (ln == 0) redbuf[w][r] = s;
        }
    }
    __syncthreads();
    if (threadIdx.x < 32) {
        float s = redbuf[0][threadIdx.x];
#pragma unroll
        for (int ww = 1; ww < 8; ww++) s += redbuf[ww][threadIdx.x];
        rowsum[threadIdx.x] = s;
    }
    __syncthreads();

    // ---- phase 3: one 16x16 ctx tile per wave over K=512
    const int mi3 = w & 1;
    const int n3  = (w >> 1) * 16;
    const half_t* vb    = VcT + ((size_t)bh * HD + n3 + ln) * TOPK + hf * 8;
    const half_t* pbase = &probs[mi3 * 16 + ln][hf * 8];

    v8f acc = {};
    for (int kk = 0; kk < TOPK; kk += 32) {
        __builtin_prefetch(vb + kk + 64, 0, 0);   // stream hint on V operand
        v16h pa  = frag_f16(pbase + kk);
        v16h vbf = frag_f16(vb + kk);
        acc = wmma16(pa, vbf, acc);
    }

#pragma unroll
    for (int v = 0; v < 8; v++) {
        int r = mi3 * 16 + v + 8 * hf;
        int s = q0 + r;
        float val = acc[v] / rowsum[r];
        ctx16[((size_t)b * SS + s) * DD + h * HD + n3 + ln] = (half_t)val;
    }
}

// ---------------------------------------------------------------------------
// Launch
// ---------------------------------------------------------------------------
extern "C" void kernel_launch(void* const* d_in, const int* in_sizes, int n_in,
                              void* d_out, int out_size, void* d_ws, size_t ws_size,
                              hipStream_t stream) {
    (void)in_sizes; (void)n_in; (void)out_size; (void)ws_size;

    const float* query = (const float*)d_in[0];
    const float* key   = (const float*)d_in[1];
    const float* value = (const float*)d_in[2];
    const float* mask  = (const float*)d_in[3];
    const unsigned char* uset = (const unsigned char*)d_in[4];
    const float* Wq = (const float*)d_in[5];
    const float* bq = (const float*)d_in[6];
    const float* Wk = (const float*)d_in[7];
    const float* bk = (const float*)d_in[8];
    const float* Wv = (const float*)d_in[9];
    const float* bv = (const float*)d_in[10];
    const float* Wo = (const float*)d_in[11];
    const float* bo = (const float*)d_in[12];
    float* out = (float*)d_out;

    // workspace layout
    char* ws = (char*)d_ws;
    const size_t szW = (size_t)DD * DD * sizeof(half_t);
    half_t* Wq16 = (half_t*)(ws);
    half_t* Wk16 = (half_t*)(ws + szW);
    half_t* Wv16 = (half_t*)(ws + 2 * szW);
    half_t* Wo16 = (half_t*)(ws + 3 * szW);
    size_t off = 4 * szW;
    int*    idx     = (int*)(ws + off);           off += TOPK * sizeof(int);
    off = (off + 255) & ~(size_t)255;
    float*  keybias = (float*)(ws + off);         off += (size_t)NB * TOPK * sizeof(float);
    off = (off + 255) & ~(size_t)255;
    half_t* q16x  = (half_t*)(ws + off);          off += (size_t)NB * SS * DD * sizeof(half_t);
    half_t* kg16  = (half_t*)(ws + off);          off += (size_t)NB * TOPK * DD * sizeof(half_t);
    half_t* vg16  = (half_t*)(ws + off);          off += (size_t)NB * TOPK * DD * sizeof(half_t);
    half_t* Q16   = (half_t*)(ws + off);          off += (size_t)NB * HH * SS * HD * sizeof(half_t);
    half_t* Kc16  = (half_t*)(ws + off);          off += (size_t)NB * HH * TOPK * HD * sizeof(half_t);
    half_t* VcT   = (half_t*)(ws + off);          off += (size_t)NB * HH * HD * TOPK * sizeof(half_t);
    half_t* ctx16 = (half_t*)(ws + off);          off += (size_t)NB * SS * DD * sizeof(half_t);

    // 1) weights -> f16
    cvt_weights_kernel<<<(DD * DD + 255) / 256, 256, 0, stream>>>(
        Wq, Wk, Wv, Wo, Wq16, Wk16, Wv16, Wo16);

    // 2) compact active key indices + per-batch key bias
    build_idx_kernel<<<1, 256, 0, stream>>>(uset, mask, idx, keybias);

    // 3) activations -> f16 (key/value: gathered rows only, 4x less work)
    cvt_rows_kernel<<<NB * SS, 192, 0, stream>>>(query, idx, q16x, 0);
    cvt_rows_kernel<<<NB * TOPK, 192, 0, stream>>>(key, idx, kg16, 1);
    cvt_rows_kernel<<<NB * TOPK, 192, 0, stream>>>(value, idx, vg16, 1);

    // 4-6) projections (pure f16 WMMA GEMMs)
    gemm_nt_kernel<<<dim3(NB * SS / 64, DD / 128), 128, 0, stream>>>(
        q16x, Wq16, bq, Q16, nullptr, 0, 0.125f);
    gemm_nt_kernel<<<dim3(NB * TOPK / 64, DD / 128), 128, 0, stream>>>(
        kg16, Wk16, bk, Kc16, nullptr, 1, 1.0f);
    gemm_nt_kernel<<<dim3(NB * TOPK / 64, DD / 128), 128, 0, stream>>>(
        vg16, Wv16, bv, VcT, nullptr, 2, 1.0f);

    // 7) attention over 512 keys
    attn_kernel<<<dim3(SS / 32, NB * HH), 256, 0, stream>>>(
        Q16, Kc16, VcT, keybias, ctx16);

    // 8) output projection -> f32
    gemm_nt_kernel<<<dim3(NB * SS / 64, DD / 128), 128, 0, stream>>>(
        ctx16, Wo16, bo, nullptr, out, 3, 1.0f);
}
